// TbNet_54726473285735
// MI455X (gfx1250) — matrix operand order repaired and checked by reading, not compiled, not run on hardware
//
#include <hip/hip_runtime.h>
#include <hip/hip_bf16.h>

#define NN 50000
#define EE 800000
#define TT 16

typedef __attribute__((ext_vector_type(16))) _Float16 v16h;
typedef __attribute__((ext_vector_type(8)))  _Float16 v8h;
typedef __attribute__((ext_vector_type(4)))  _Float16 v4h;
typedef __attribute__((ext_vector_type(8)))  float    v8f;

__device__ __forceinline__ v8f wmma_acc(v16h a, v16h b, v8f c) {
  // D = A(16x32 f16) * B(32x16 f16) + C(16x16 f32)
  return __builtin_amdgcn_wmma_f32_16x16x32_f16(false, a, false, b, (short)0, c, false, false);
}

// B-fragment from LDS laid out as [n][k] with f16 stride `stride`:
// lane gives N=nrow, element e holds K = kt*32 + hi*8 + e (e<8), kt*32 + 16 + hi*8 + (e-8) (e>=8).
__device__ __forceinline__ v16h ldsB(const _Float16* __restrict__ base, int nrow,
                                     int kt, int hi, int stride) {
  const _Float16* p = base + nrow * stride + kt * 32 + hi * 8;
  v16h B;
#pragma unroll
  for (int i = 0; i < 4; ++i) {
    v4h q = *(const v4h*)(p + (i & 1) * 4 + (i >> 1) * 16);
    B[4 * i + 0] = q[0]; B[4 * i + 1] = q[1];
    B[4 * i + 2] = q[2]; B[4 * i + 3] = q[3];
  }
  return B;
}

__device__ __forceinline__ float sigmoidf_(float x) { return 1.f / (1.f + __expf(-x)); }
__device__ __forceinline__ float tanh_fast(float x) {
  float e = __expf(2.f * x);
  return 1.f - 2.f / (e + 1.f);   // stable: +inf -> 1, 0 -> -1
}

// ---------------------------------------------------------------------------
// 1) Embedding gather + GRU (16 steps) -> hcat[N][96] f16 = [x(8) | hT(64) | 0(24)]
// ---------------------------------------------------------------------------
__global__ void __launch_bounds__(128)
gru_hcat_kernel(const float* __restrict__ x, const int* __restrict__ xtext,
                const float* __restrict__ embed,
                const float* __restrict__ wih, const float* __restrict__ whh,
                const float* __restrict__ bih, const float* __restrict__ bhh,
                _Float16* __restrict__ hcat) {
  __shared__ __align__(16) _Float16 sWih[192 * 68];
  __shared__ __align__(16) _Float16 sWhh[192 * 68];
  __shared__ __align__(16) _Float16 sH[4][16 * 72];

  const int tid = threadIdx.x;
  for (int idx = tid; idx < 192 * 64; idx += 128) {
    int n = idx >> 6, k = idx & 63;
    sWih[n * 68 + k] = (_Float16)wih[idx];   // B[k][n] = w[n][k] (gi = x @ w^T)
    sWhh[n * 68 + k] = (_Float16)whh[idx];
  }
  __syncthreads();

  const int wave = tid >> 5, lane = tid & 31, l15 = lane & 15, hi = lane >> 4;
  const int tile = blockIdx.x * 4 + wave;
  if (tile >= NN / 16) return;
  const int node = tile * 16 + l15;
  const int klo = hi * 8;
  _Float16* sh = &sH[wave][0];

  // per-lane gate biases: column = jj*16 + l15 in each 64-wide gate block
  float br[4], bz[4], bi_n[4], bh_n[4];
#pragma unroll
  for (int jj = 0; jj < 4; ++jj) {
    int c = jj * 16 + l15;
    br[jj]   = bih[c] + bhh[c];
    bz[jj]   = bih[64 + c] + bhh[64 + c];
    bi_n[jj] = bih[128 + c];
    bh_n[jj] = bhh[128 + c];
  }

  v16h Ah0 = {}; v16h Ah1 = {};    // h0 = 0
  float Hc[4][8];
#pragma unroll
  for (int jj = 0; jj < 4; ++jj)
#pragma unroll
    for (int r = 0; r < 8; ++r) Hc[jj][r] = 0.f;

  for (int t = 0; t < TT; ++t) {
    // ---- A fragments for x_t from embedding gather (row M = l15) ----
    const int widx = xtext[node * TT + t];
    const float* row = embed + (size_t)widx * 64;
    v16h Ax0, Ax1;
#pragma unroll
    for (int kt = 0; kt < 2; ++kt) {
      const float* p = row + kt * 32 + klo;
      float4 c0 = *(const float4*)(p);
      float4 c1 = *(const float4*)(p + 4);
      float4 c2 = *(const float4*)(p + 16);
      float4 c3 = *(const float4*)(p + 20);
      v16h A;
      A[0]  = (_Float16)c0.x; A[1]  = (_Float16)c0.y; A[2]  = (_Float16)c0.z; A[3]  = (_Float16)c0.w;
      A[4]  = (_Float16)c1.x; A[5]  = (_Float16)c1.y; A[6]  = (_Float16)c1.z; A[7]  = (_Float16)c1.w;
      A[8]  = (_Float16)c2.x; A[9]  = (_Float16)c2.y; A[10] = (_Float16)c2.z; A[11] = (_Float16)c2.w;
      A[12] = (_Float16)c3.x; A[13] = (_Float16)c3.y; A[14] = (_Float16)c3.z; A[15] = (_Float16)c3.w;
      if (kt == 0) Ax0 = A; else Ax1 = A;
    }

    // ---- gates per 16-column group jj ----
#pragma unroll
    for (int jj = 0; jj < 4; ++jj) {
      v8f cir = {}; v8f chr = {}; v8f ciz = {}; v8f chz = {}; v8f cin = {}; v8f chn = {};
      cir = wmma_acc(Ax0, ldsB(sWih, jj * 16 + l15, 0, hi, 68), cir);
      cir = wmma_acc(Ax1, ldsB(sWih, jj * 16 + l15, 1, hi, 68), cir);
      chr = wmma_acc(Ah0, ldsB(sWhh, jj * 16 + l15, 0, hi, 68), chr);
      chr = wmma_acc(Ah1, ldsB(sWhh, jj * 16 + l15, 1, hi, 68), chr);
      ciz = wmma_acc(Ax0, ldsB(sWih, (jj + 4) * 16 + l15, 0, hi, 68), ciz);
      ciz = wmma_acc(Ax1, ldsB(sWih, (jj + 4) * 16 + l15, 1, hi, 68), ciz);
      chz = wmma_acc(Ah0, ldsB(sWhh, (jj + 4) * 16 + l15, 0, hi, 68), chz);
      chz = wmma_acc(Ah1, ldsB(sWhh, (jj + 4) * 16 + l15, 1, hi, 68), chz);
      cin = wmma_acc(Ax0, ldsB(sWih, (jj + 8) * 16 + l15, 0, hi, 68), cin);
      cin = wmma_acc(Ax1, ldsB(sWih, (jj + 8) * 16 + l15, 1, hi, 68), cin);
      chn = wmma_acc(Ah0, ldsB(sWhh, (jj + 8) * 16 + l15, 0, hi, 68), chn);
      chn = wmma_acc(Ah1, ldsB(sWhh, (jj + 8) * 16 + l15, 1, hi, 68), chn);
#pragma unroll
      for (int r = 0; r < 8; ++r) {
        float rg = sigmoidf_(cir[r] + chr[r] + br[jj]);
        float zg = sigmoidf_(ciz[r] + chz[r] + bz[jj]);
        float ng = tanh_fast(cin[r] + bi_n[jj] + rg * (chn[r] + bh_n[jj]));
        float hn = (1.f - zg) * ng + zg * Hc[jj][r];
        Hc[jj][r] = hn;
        // transpose staging: row M = r+8*hi, col N = jj*16+l15
        sh[(r + 8 * hi) * 72 + jj * 16 + l15] = (_Float16)hn;
      }
    }
    asm volatile("s_wait_dscnt 0" ::: "memory");
    // rebuild A fragments of h for next step's recurrent GEMM
#pragma unroll
    for (int kt = 0; kt < 2; ++kt) {
      const _Float16* p = sh + l15 * 72 + kt * 32 + klo;
      v8h c1 = *(const v8h*)(p);
      v8h c2 = *(const v8h*)(p + 16);
      v16h A;
#pragma unroll
      for (int i = 0; i < 8; ++i) { A[i] = c1[i]; A[8 + i] = c2[i]; }
      if (kt == 0) Ah0 = A; else Ah1 = A;
    }
  }

  // ---- epilogue: hcat[node] = [x(8) | hT(64) | zeros(24)] as f16 ----
  {
    const int cbase = hi * 32;
    _Float16* dsth = hcat + (size_t)node * 96 + 8 + cbase;
    const _Float16* srcp = sh + l15 * 72 + cbase;
#pragma unroll
    for (int i = 0; i < 4; ++i)
      *(v8h*)(dsth + i * 8) = *(const v8h*)(srcp + i * 8);
  }
  if (hi == 0) {
    float4 a = *(const float4*)(x + (size_t)node * 8);
    float4 b = *(const float4*)(x + (size_t)node * 8 + 4);
    v8h xv;
    xv[0] = (_Float16)a.x; xv[1] = (_Float16)a.y; xv[2] = (_Float16)a.z; xv[3] = (_Float16)a.w;
    xv[4] = (_Float16)b.x; xv[5] = (_Float16)b.y; xv[6] = (_Float16)b.z; xv[7] = (_Float16)b.w;
    *(v8h*)(hcat + (size_t)node * 96) = xv;
  } else {
    v8h z = {};
#pragma unroll
    for (int i = 0; i < 3; ++i)
      *(v8h*)(hcat + (size_t)node * 96 + 72 + i * 8) = z;
  }
}

// ---------------------------------------------------------------------------
// 2) degree / dinv
// ---------------------------------------------------------------------------
__global__ void deg_init_kernel(float* __restrict__ deg, int n) {
  int i = blockIdx.x * blockDim.x + threadIdx.x;
  if (i < n) deg[i] = 1.f;            // self loop
}
__global__ void deg_count_kernel(const int* __restrict__ col, float* __restrict__ deg, int e) {
  int i = blockIdx.x * blockDim.x + threadIdx.x;
  if (i < e) atomicAdd(&deg[col[i]], 1.f);
}
__global__ void deg_finish_kernel(float* __restrict__ deg, int n) {
  int i = blockIdx.x * blockDim.x + threadIdx.x;
  if (i < n) deg[i] = rsqrtf(fmaxf(deg[i], 1.f));
}

// ---------------------------------------------------------------------------
// 3) GEMM: M[N][64] = Ain_f16[N][kstride] @ W[kin][64]   (zero-padded K)
// ---------------------------------------------------------------------------
__global__ void __launch_bounds__(128)
gcn_mm_kernel(const _Float16* __restrict__ Ain, int kstride, int nkt,
              const float* __restrict__ W, int kin,
              float* __restrict__ Mout, int ntiles) {
  __shared__ __align__(16) _Float16 sW[64 * 100];   // transposed [n][k], k < nkt*32
  const int tid = threadIdx.x;
  const int kpad = nkt * 32;
  for (int idx = tid; idx < 64 * kpad; idx += 128) {
    int n = idx / kpad, k = idx - n * kpad;
    float v = (k < kin) ? W[k * 64 + n] : 0.f;
    sW[n * 100 + k] = (_Float16)v;
  }
  __syncthreads();

  const int wave = tid >> 5, lane = tid & 31, l15 = lane & 15, hi = lane >> 4;
  const int tile = blockIdx.x * 4 + wave;
  if (tile >= ntiles) return;
  const int node = tile * 16 + l15;
  const int klo = hi * 8;

  v16h A[3];
  for (int kt = 0; kt < nkt; ++kt) {
    const _Float16* p = Ain + (size_t)node * kstride + kt * 32 + klo;
    v8h c1 = *(const v8h*)(p);
    v8h c2 = *(const v8h*)(p + 16);
#pragma unroll
    for (int i = 0; i < 8; ++i) { A[kt][i] = c1[i]; A[kt][8 + i] = c2[i]; }
  }
#pragma unroll
  for (int nt = 0; nt < 4; ++nt) {
    v8f acc = {};
    for (int kt = 0; kt < nkt; ++kt)
      acc = wmma_acc(A[kt], ldsB(sW, nt * 16 + l15, kt, hi, 100), acc);
#pragma unroll
    for (int r = 0; r < 8; ++r) {
      int rowg = tile * 16 + r + 8 * hi;
      Mout[(size_t)rowg * 64 + nt * 16 + l15] = acc[r];
    }
  }
}

// ---------------------------------------------------------------------------
// 4) scatter (segment-sum over edges), 16 threads / edge, float4 per thread
// ---------------------------------------------------------------------------
__global__ void zero_kernel(float* __restrict__ p, int n) {
  int i = blockIdx.x * blockDim.x + threadIdx.x;
  if (i < n) p[i] = 0.f;
}
__global__ void scatter_kernel(const int* __restrict__ e0, const int* __restrict__ e1,
                               const float* __restrict__ dinv, const float* __restrict__ m,
                               float* __restrict__ agg) {
  int tid = blockIdx.x * blockDim.x + threadIdx.x;
  if (tid >= EE * 16) return;
  int e = tid >> 4;
  int c = (tid & 15) << 2;
  int r = e0[e], col = e1[e];
  float nrm = dinv[r] * dinv[col];
  float4 v = *(const float4*)(m + (size_t)r * 64 + c);
  float* dst = agg + (size_t)col * 64 + c;
  atomicAdd(dst + 0, nrm * v.x);
  atomicAdd(dst + 1, nrm * v.y);
  atomicAdd(dst + 2, nrm * v.z);
  atomicAdd(dst + 3, nrm * v.w);
}

// 5) h = relu(agg + dinv^2 * m + b), store as f16 [N][64]
__global__ void finalize_kernel(const float* __restrict__ m, const float* __restrict__ agg,
                                const float* __restrict__ dinv, const float* __restrict__ b,
                                _Float16* __restrict__ hout) {
  int tid = blockIdx.x * blockDim.x + threadIdx.x;
  if (tid >= NN * 16) return;
  int v = tid >> 4, c = (tid & 15) << 2;
  float d2 = dinv[v] * dinv[v];
  float4 mm = *(const float4*)(m + (size_t)v * 64 + c);
  float4 ag = *(const float4*)(agg + (size_t)v * 64 + c);
  float4 bb = *(const float4*)(b + c);
  v4h o;
  o[0] = (_Float16)fmaxf(ag.x + d2 * mm.x + bb.x, 0.f);
  o[1] = (_Float16)fmaxf(ag.y + d2 * mm.y + bb.y, 0.f);
  o[2] = (_Float16)fmaxf(ag.z + d2 * mm.z + bb.z, 0.f);
  o[3] = (_Float16)fmaxf(ag.w + d2 * mm.w + bb.w, 0.f);
  *(v4h*)(hout + (size_t)v * 64 + c) = o;
}

// ---------------------------------------------------------------------------
// 6) edge classifier: [h2[src] | h2[dst]](16x128) @ Wl1 -> relu -> @Wf -> log_softmax
// ---------------------------------------------------------------------------
__global__ void __launch_bounds__(128)
edge_cls_kernel(const int* __restrict__ e0, const int* __restrict__ e1,
                const _Float16* __restrict__ h, const float* __restrict__ Wl1,
                const float* __restrict__ bl1, const float* __restrict__ Wf,
                const float* __restrict__ bf, float* __restrict__ out) {
  __shared__ __align__(16) _Float16 sW[64 * 132];   // Wl1 transposed [n][k], k<128
  const int tid = threadIdx.x;
  for (int idx = tid; idx < 64 * 128; idx += 128) {
    int n = idx >> 7, k = idx & 127;
    sW[n * 132 + k] = (_Float16)Wl1[k * 64 + n];
  }
  __syncthreads();

  const int wave = tid >> 5, lane = tid & 31, l15 = lane & 15, hi = lane >> 4;
  const int tile = blockIdx.x * 4 + wave;
  if (tile >= EE / 16) return;
  const int base = tile * 16;
  const int klo = hi * 8;

  const int eidx = base + l15;
  const int src = e0[eidx], dst = e1[eidx];

  v16h A[4];
#pragma unroll
  for (int kt = 0; kt < 4; ++kt) {
    const _Float16* hr = h + (size_t)((kt < 2) ? src : dst) * 64 + (kt & 1) * 32 + klo;
    v8h c1 = *(const v8h*)(hr);
    v8h c2 = *(const v8h*)(hr + 16);
#pragma unroll
    for (int i = 0; i < 8; ++i) { A[kt][i] = c1[i]; A[kt][8 + i] = c2[i]; }
  }

  float blv[4], wf0[4], wf1[4];
#pragma unroll
  for (int nt = 0; nt < 4; ++nt) {
    int c = nt * 16 + l15;
    blv[nt] = bl1[c];
    wf0[nt] = Wf[2 * c];
    wf1[nt] = Wf[2 * c + 1];
  }

  float p0[8], p1[8];
#pragma unroll
  for (int r = 0; r < 8; ++r) { p0[r] = 0.f; p1[r] = 0.f; }

#pragma unroll
  for (int nt = 0; nt < 4; ++nt) {
    v8f acc = {};
#pragma unroll
    for (int kt = 0; kt < 4; ++kt)
      acc = wmma_acc(A[kt], ldsB(sW, nt * 16 + l15, kt, hi, 132), acc);
#pragma unroll
    for (int r = 0; r < 8; ++r) {
      float hm = fmaxf(acc[r] + blv[nt], 0.f);
      p0[r] += hm * wf0[nt];
      p1[r] += hm * wf1[nt];
    }
  }
  // butterfly sum over the 16-lane column group (rows fixed per half-wave)
#pragma unroll
  for (int off = 1; off < 16; off <<= 1) {
#pragma unroll
    for (int r = 0; r < 8; ++r) {
      p0[r] += __shfl_xor(p0[r], off, 32);
      p1[r] += __shfl_xor(p1[r], off, 32);
    }
  }
  if (l15 == 0) {
    float bf0 = bf[0], bf1 = bf[1];
#pragma unroll
    for (int r = 0; r < 8; ++r) {
      int rowg = base + r + 8 * hi;
      float l0 = p0[r] + bf0, l1 = p1[r] + bf1;
      float mx = fmaxf(l0, l1);
      float lse = mx + __logf(__expf(l0 - mx) + __expf(l1 - mx));
      out[(size_t)rowg * 2 + 0] = l0 - lse;
      out[(size_t)rowg * 2 + 1] = l1 - lse;
    }
  }
}

// ---------------------------------------------------------------------------
extern "C" void kernel_launch(void* const* d_in, const int* in_sizes, int n_in,
                              void* d_out, int out_size, void* d_ws, size_t ws_size,
                              hipStream_t stream) {
  (void)in_sizes; (void)n_in; (void)out_size; (void)ws_size;
  const float* x     = (const float*)d_in[0];
  const int*   ei    = (const int*)d_in[1];
  const int*   e0    = ei;
  const int*   e1    = ei + EE;
  const int*   xtext = (const int*)d_in[2];
  const float* embed = (const float*)d_in[3];
  const float* wih   = (const float*)d_in[4];
  const float* whh   = (const float*)d_in[5];
  const float* bih   = (const float*)d_in[6];
  const float* bhh   = (const float*)d_in[7];
  const float* W1    = (const float*)d_in[8];
  const float* b1    = (const float*)d_in[9];
  const float* W2    = (const float*)d_in[10];
  const float* b2    = (const float*)d_in[11];
  const float* Wl1   = (const float*)d_in[12];
  const float* bl1   = (const float*)d_in[13];
  const float* Wf    = (const float*)d_in[14];
  const float* bf    = (const float*)d_in[15];
  float* out = (float*)d_out;

  char* ws = (char*)d_ws;
  _Float16* hcat = (_Float16*)(ws);                 //  9,600,000 B : [N][96] f16
  _Float16* h16  = (_Float16*)(ws + 9600000);       //  6,400,000 B : [N][64] f16
  float*    mbuf = (float*)(ws + 16000000);         // 12,800,000 B : [N][64] f32
  float*    agg  = (float*)(ws + 28800000);         // 12,800,000 B : [N][64] f32
  float*    dinv = (float*)(ws + 41600000);         //    200,000 B : [N]     f32

  const int ntiles = NN / 16;                       // 3125
  const int gemmBlocks = (ntiles + 3) / 4;          // 4 waves/block

  // text encoder
  gru_hcat_kernel<<<gemmBlocks, 128, 0, stream>>>(x, xtext, embed, wih, whh, bih, bhh, hcat);

  // symmetric GCN normalization
  deg_init_kernel<<<(NN + 255) / 256, 256, 0, stream>>>(dinv, NN);
  deg_count_kernel<<<(EE + 255) / 256, 256, 0, stream>>>(e1, dinv, EE);
  deg_finish_kernel<<<(NN + 255) / 256, 256, 0, stream>>>(dinv, NN);

  // GCN layer 1: hcat[N][96] @ W1(72x64, zero-padded)
  gcn_mm_kernel<<<gemmBlocks, 128, 0, stream>>>(hcat, 96, 3, W1, 72, mbuf, ntiles);
  zero_kernel<<<(NN * 64 + 255) / 256, 256, 0, stream>>>(agg, NN * 64);
  scatter_kernel<<<(EE * 16 + 255) / 256, 256, 0, stream>>>(e0, e1, dinv, mbuf, agg);
  finalize_kernel<<<(NN * 16 + 255) / 256, 256, 0, stream>>>(mbuf, agg, dinv, b1, h16);

  // GCN layer 2: h1[N][64] @ W2(64x64)
  gcn_mm_kernel<<<gemmBlocks, 128, 0, stream>>>(h16, 64, 2, W2, 64, mbuf, ntiles);
  zero_kernel<<<(NN * 64 + 255) / 256, 256, 0, stream>>>(agg, NN * 64);
  scatter_kernel<<<(EE * 16 + 255) / 256, 256, 0, stream>>>(e0, e1, dinv, mbuf, agg);
  finalize_kernel<<<(NN * 16 + 255) / 256, 256, 0, stream>>>(mbuf, agg, dinv, b2, h16);

  // edge-pair classifier + log_softmax
  edge_cls_kernel<<<(EE / 16 + 3) / 4, 128, 0, stream>>>(e0, e1, h16, Wl1, bl1, Wf, bf, out);
}
